// MagneticMACE_2903397893055
// MI455X (gfx1250) — matrix operand order repaired and checked by reading, not compiled, not run on hardware
//
#include <hip/hip_runtime.h>
#include <hip/hip_bf16.h>
#include <math.h>

#define N_NODES 10000
#define N_EDGES 160000
#define NF 32
#define NC 8
#define WAVES 4
#define PI_F 3.14159265358979323846f

typedef __attribute__((ext_vector_type(16))) __bf16 v16bf;
typedef __attribute__((ext_vector_type(8)))  float  v8f;

union Frag { v16bf v; unsigned int u[8]; unsigned short h[16]; };

// round-to-nearest-even f32 -> bf16 (used only for one-time weight staging)
__device__ __forceinline__ unsigned short f2bh(float f) {
  unsigned u = __float_as_uint(f);
  return (unsigned short)((u + 0x7FFFu + ((u >> 16) & 1u)) >> 16);
}
// truncating f32 -> bf16: lowers to ds_store_b16_d16_hi (zero ALU cost)
__device__ __forceinline__ unsigned short f2bh_t(float f) {
  return (unsigned short)(__float_as_uint(f) >> 16);
}
__device__ __forceinline__ float bh2f(unsigned short h) {
  return __uint_as_float(((unsigned)h) << 16);
}
// silu with fast v_rcp_f32 instead of IEEE division sequence
__device__ __forceinline__ float silu_f(float x) {
  return x * __builtin_amdgcn_rcpf(1.0f + __expf(-x));
}

__global__ void zero_kernel(float* __restrict__ out) {
  int i = blockIdx.x * blockDim.x + threadIdx.x;
  if (i < N_NODES) out[i] = 0.0f;
}

// Per-node precompute: cheb [N,8], mY [N,16 (9 used)], h0 [N,32],
// hw = Wread*h0 [N,32], gw = Wread*silu(cheb@Wmag) [N,32]
__global__ void node_kernel(const float* __restrict__ magmoms,
                            const int*   __restrict__ species,
                            const float* __restrict__ W_embed,
                            const float* __restrict__ Wmag,
                            const float* __restrict__ Wread,
                            float* __restrict__ cheb_o,
                            float* __restrict__ mY_o,
                            float* __restrict__ h0_o,
                            float* __restrict__ hw_o,
                            float* __restrict__ gw_o) {
  int n = blockIdx.x * blockDim.x + threadIdx.x;
  if (n >= N_NODES) return;
  float mx = magmoms[3*n+0], my = magmoms[3*n+1], mz = magmoms[3*n+2];
  float mn = sqrtf(mx*mx + my*my + mz*mz + 1e-12f);
  float xt = tanhf(mn);
  float T[NC];
  { float Tp = 1.0f, Tc = xt; T[0] = Tc;
    for (int k = 1; k < NC; ++k) { float Tn = 2.0f*xt*Tc - Tp; T[k] = Tn; Tp = Tc; Tc = Tn; } }
  for (int k = 0; k < NC; ++k) cheb_o[n*NC + k] = T[k];
  float inv = 1.0f / mn;
  float x = mx*inv, y = my*inv, z = mz*inv;
  const float s3 = 1.7320508075688772f, s15 = 3.872983346207417f, s5 = 2.23606797749979f;
  float Y[9];
  Y[0] = 1.0f; Y[1] = s3*x; Y[2] = s3*y; Y[3] = s3*z;
  Y[4] = s15*x*y; Y[5] = s15*y*z; Y[6] = 0.5f*s5*(3.0f*z*z - 1.0f);
  Y[7] = s15*x*z; Y[8] = 0.5f*s15*(x*x - y*y);
  for (int k = 0; k < 9; ++k) mY_o[n*16 + k] = Y[k];
  int sp = species[n];
  for (int f = 0; f < NF; ++f) {
    float acc = 0.0f;
    for (int k = 0; k < NC; ++k) acc += T[k] * Wmag[k*NF + f];
    float wr = Wread[f];
    gw_o[n*NF + f] = silu_f(acc) * wr;
    float h = W_embed[sp*NF + f];
    h0_o[n*NF + f] = h;
    hw_o[n*NF + f] = h * wr;
  }
}

__global__ void __launch_bounds__(128)
edge_kernel(const float* __restrict__ positions,
            const int*   __restrict__ senders,
            const int*   __restrict__ receivers,
            const float* __restrict__ W0g, const float* __restrict__ b0g,
            const float* __restrict__ W1g, const float* __restrict__ b1g,
            const float* __restrict__ W2g, const float* __restrict__ b2g,
            const float* __restrict__ cheb_n, const float* __restrict__ mY_n,
            const float* __restrict__ h0_n,  const float* __restrict__ hw_n,
            const float* __restrict__ gw_n,
            float* __restrict__ out) {
  // Weights pre-swizzled into WMMA B-fragment layout (bf16, 8 uints/lane contiguous)
  __shared__ unsigned int s_bsw0[4  * 256];
  __shared__ unsigned int s_bsw1[8  * 256];
  __shared__ unsigned int s_bsw2[16 * 256];
  __shared__ float s_b1[64], s_b2[128];
  __shared__ unsigned int s_actA[WAVES][512];   // 16x64 bf16
  __shared__ unsigned int s_actB[WAVES][512];   // 16x64 bf16
  __shared__ unsigned int s_wout[WAVES][1024];  // 16x128 bf16; first 256 uints double as rad_in buf

  const int tid  = threadIdx.x;
  const int wave = tid >> 5;
  const int l    = tid & 31;
  const int half = l >> 4;   // lane half (wave32 WMMA layout)
  const int lr   = l & 15;

  // ---- stage weights into swizzled bf16 fragments ----
  {
    unsigned short* h0p = (unsigned short*)s_bsw0;
    for (int idx = tid; idx < 4*512; idx += 128) {
      int frag = idx >> 9, rem = idx & 511, ln = rem >> 4, i = rem & 15;
      int n = frag*16 + (ln & 15);
      int k = ((ln >> 4) << 3) + (i & 7) + ((i >> 3) << 4);            // 0..31; k==16 row = bias (A row 16 is 1.0)
      float v = (k < 16) ? W0g[k*64 + n] : ((k == 16) ? b0g[n] : 0.0f);
      h0p[idx] = f2bh(v);
    }
    unsigned short* h1p = (unsigned short*)s_bsw1;
    for (int idx = tid; idx < 8*512; idx += 128) {
      int frag = idx >> 9, rem = idx & 511, ln = rem >> 4, i = rem & 15;
      int nt = frag >> 1, ks = frag & 1;
      int n = nt*16 + (ln & 15);
      int k = ks*32 + ((ln >> 4) << 3) + (i & 7) + ((i >> 3) << 4);
      h1p[idx] = f2bh(W1g[k*64 + n]);
    }
    unsigned short* h2p = (unsigned short*)s_bsw2;
    for (int idx = tid; idx < 16*512; idx += 128) {
      int frag = idx >> 9, rem = idx & 511, ln = rem >> 4, i = rem & 15;
      int nt = frag >> 1, ks = frag & 1;
      int n = nt*16 + (ln & 15);
      int k = ks*32 + ((ln >> 4) << 3) + (i & 7) + ((i >> 3) << 4);
      h2p[idx] = f2bh(W2g[k*128 + n]);
    }
    for (int i = tid; i < 64; i += 128) s_b1[i] = b1g[i];
    for (int i = tid; i < 128; i += 128) s_b2[i] = b2g[i];
  }

  // ---- per-edge setup (both lane halves compute the same edge scalars) ----
  const int e = (blockIdx.x * WAVES + wave) * 16 + lr;   // N_EDGES % 64 == 0
  const int snd = senders[e];
  const int rcv = receivers[e];

  float dx = positions[3*rcv+0] - positions[3*snd+0];
  float dy = positions[3*rcv+1] - positions[3*snd+1];
  float dz = positions[3*rcv+2] - positions[3*snd+2];
  float rl = sqrtf(dx*dx + dy*dy + dz*dz + 1e-12f);
  float ir = __builtin_amdgcn_rcpf(rl);
  float ux = dx*ir, uy = dy*ir, uz = dz*ir;
  const float s3 = 1.7320508075688772f, s15 = 3.872983346207417f, s5 = 2.23606797749979f;
  float Y1 = s3*ux, Y2 = s3*uy, Y3 = s3*uz;
  float Y4 = s15*ux*uy, Y5 = s15*uy*uz, Y6 = 0.5f*s5*(3.0f*uz*uz - 1.0f);
  float Y7 = s15*ux*uz, Y8 = 0.5f*s15*(ux*ux - uy*uy);
  const float* mYr = mY_n + rcv*16;
  float t0 = mYr[0];
  float t1 = Y1*mYr[1] + Y2*mYr[2] + Y3*mYr[3];
  float t2 = Y4*mYr[4] + Y5*mYr[5] + Y6*mYr[6] + Y7*mYr[7] + Y8*mYr[8];

  // Bessel * polynomial cutoff (p=5), then Chebyshev(sender): rad_in[16]; rad_in[16]=1 (bias row)
  float xs = rl * 0.2f;                  // r / R_MAX
  float fc = 0.0f;
  if (xs < 1.0f) {
    float x2 = xs*xs; float x5 = x2*x2*xs;
    fc = 1.0f - 21.0f*x5 + 35.0f*x5*xs - 15.0f*x5*x2;
  }
  float scale = 0.6324555320336759f * ir * fc;  // sqrt(2/R_MAX)/r * fc
  unsigned short* radh = (unsigned short*)s_wout[wave];
  if (half == 0) {
    #pragma unroll
    for (int nb = 1; nb <= 8; ++nb)
      radh[lr*32 + (nb-1)] = f2bh_t(__sinf((float)nb * PI_F * xs) * scale);
    #pragma unroll
    for (int k = 0; k < 8; ++k)
      radh[lr*32 + 8 + k] = f2bh_t(cheb_n[snd*8 + k]);
    radh[lr*32 + 16] = 0x3F80;           // bf16 1.0 -> multiplies staged b0 row
    #pragma unroll
    for (int k = 17; k < 32; ++k) radh[lr*32 + k] = 0;
  }
  __syncthreads();

  // ---- layer 0: (16x32) @ (32x64), K padded, bias folded in ----
  const unsigned int* radu = s_wout[wave];
  Frag a0;
  #pragma unroll
  for (int j = 0; j < 4; ++j) {
    a0.u[j]     = radu[lr*16 + half*4 + j];
    a0.u[4 + j] = radu[lr*16 + 8 + half*4 + j];
  }
  unsigned short* actAh = (unsigned short*)s_actA[wave];
  #pragma unroll
  for (int nt = 0; nt < 4; ++nt) {
    Frag b;
    #pragma unroll
    for (int j = 0; j < 8; ++j) b.u[j] = s_bsw0[nt*256 + l*8 + j];
    v8f c = {};
    c = __builtin_amdgcn_wmma_f32_16x16x32_bf16(false, a0.v, false, b.v, (short)0, c, false, false);
    int col = nt*16 + lr;
    #pragma unroll
    for (int v = 0; v < 8; ++v)
      actAh[(v + half*8)*64 + col] = f2bh_t(silu_f(c[v]));
  }
  __syncthreads();

  // ---- layer 1: (16x64) @ (64x64) ----
  const unsigned int* actAu = s_actA[wave];
  Frag a1k0, a1k1;
  #pragma unroll
  for (int j = 0; j < 4; ++j) {
    a1k0.u[j]     = actAu[lr*32 + half*4 + j];
    a1k0.u[4 + j] = actAu[lr*32 + 8 + half*4 + j];
    a1k1.u[j]     = actAu[lr*32 + 16 + half*4 + j];
    a1k1.u[4 + j] = actAu[lr*32 + 24 + half*4 + j];
  }
  unsigned short* actBh = (unsigned short*)s_actB[wave];
  #pragma unroll
  for (int nt = 0; nt < 4; ++nt) {
    Frag bf0, bf1;
    #pragma unroll
    for (int j = 0; j < 8; ++j) {
      bf0.u[j] = s_bsw1[(nt*2 + 0)*256 + l*8 + j];
      bf1.u[j] = s_bsw1[(nt*2 + 1)*256 + l*8 + j];
    }
    v8f c = {};
    c = __builtin_amdgcn_wmma_f32_16x16x32_bf16(false, a1k0.v, false, bf0.v, (short)0, c, false, false);
    c = __builtin_amdgcn_wmma_f32_16x16x32_bf16(false, a1k1.v, false, bf1.v, (short)0, c, false, false);
    int col = nt*16 + lr;
    #pragma unroll
    for (int v = 0; v < 8; ++v)
      actBh[(v + half*8)*64 + col] = f2bh_t(silu_f(c[v] + s_b1[col]));
  }
  __syncthreads();

  // ---- layer 2: (16x64) @ (64x128), bias only ----
  const unsigned int* actBu = s_actB[wave];
  Frag a2k0, a2k1;
  #pragma unroll
  for (int j = 0; j < 4; ++j) {
    a2k0.u[j]     = actBu[lr*32 + half*4 + j];
    a2k0.u[4 + j] = actBu[lr*32 + 8 + half*4 + j];
    a2k1.u[j]     = actBu[lr*32 + 16 + half*4 + j];
    a2k1.u[4 + j] = actBu[lr*32 + 24 + half*4 + j];
  }
  unsigned short* wouth = (unsigned short*)s_wout[wave];
  #pragma unroll
  for (int nt = 0; nt < 8; ++nt) {
    Frag bf0, bf1;
    #pragma unroll
    for (int j = 0; j < 8; ++j) {
      bf0.u[j] = s_bsw2[(nt*2 + 0)*256 + l*8 + j];
      bf1.u[j] = s_bsw2[(nt*2 + 1)*256 + l*8 + j];
    }
    v8f c = {};
    c = __builtin_amdgcn_wmma_f32_16x16x32_bf16(false, a2k0.v, false, bf0.v, (short)0, c, false, false);
    c = __builtin_amdgcn_wmma_f32_16x16x32_bf16(false, a2k1.v, false, bf1.v, (short)0, c, false, false);
    int col = nt*16 + lr;
    #pragma unroll
    for (int v = 0; v < 8; ++v)
      wouth[(v + half*8)*128 + col] = f2bh_t(c[v] + s_b2[col]);
  }
  __syncthreads();

  // ---- fused message + segment-sum: one scalar atomic per edge ----
  // contrib = sum_f [ hw(s,f)*w0 + gw(r,f)*h0(s,f)*(w0*t0 + w1*t1 + w2*t2) ] / 16
  const unsigned int* woutu = s_wout[wave];
  float partial = 0.0f;
  int fbase = half * 16;
  #pragma unroll 4
  for (int ff = 0; ff < 16; ++ff) {
    int f = fbase + ff;
    unsigned int u0 = woutu[lr*64 + f*2];
    unsigned int u1 = woutu[lr*64 + f*2 + 1];
    float w0v = bh2f((unsigned short)(u0 & 0xFFFFu));
    float w1v = bh2f((unsigned short)(u0 >> 16));
    float w2v = bh2f((unsigned short)(u1 & 0xFFFFu));
    float h0v = h0_n[snd*NF + f];
    float hwv = hw_n[snd*NF + f];
    float gwv = gw_n[rcv*NF + f];
    float q = w0v*t0 + w1v*t1 + w2v*t2;
    partial += hwv*w0v + gwv*h0v*q;
  }
  partial += __shfl_xor(partial, 16);
  if (half == 0)
    atomicAdd(out + rcv, partial * 0.0625f);   // / AVG_NUM_NEIGHBORS
}

extern "C" void kernel_launch(void* const* d_in, const int* in_sizes, int n_in,
                              void* d_out, int out_size, void* d_ws, size_t ws_size,
                              hipStream_t stream) {
  const float* positions = (const float*)d_in[0];
  const float* magmoms   = (const float*)d_in[1];
  const int*   species   = (const int*)  d_in[2];
  const int*   senders   = (const int*)  d_in[3];
  const int*   receivers = (const int*)  d_in[4];
  const float* W_embed   = (const float*)d_in[5];
  const float* W0        = (const float*)d_in[6];
  const float* b0        = (const float*)d_in[7];
  const float* W1        = (const float*)d_in[8];
  const float* b1        = (const float*)d_in[9];
  const float* W2        = (const float*)d_in[10];
  const float* b2        = (const float*)d_in[11];
  const float* Wmag      = (const float*)d_in[12];
  const float* Wread     = (const float*)d_in[13];

  float* out = (float*)d_out;
  float* ws  = (float*)d_ws;
  float* cheb = ws;                  // N*8
  float* mY   = ws + N_NODES*8;      // N*16
  float* h0   = ws + N_NODES*24;     // N*32
  float* hw   = ws + N_NODES*56;     // N*32
  float* gw   = ws + N_NODES*88;     // N*32  (total 4.8 MB)

  zero_kernel<<<(N_NODES + 255)/256, 256, 0, stream>>>(out);
  node_kernel<<<(N_NODES + 255)/256, 256, 0, stream>>>(
      magmoms, species, W_embed, Wmag, Wread, cheb, mY, h0, hw, gw);
  edge_kernel<<<N_EDGES/(WAVES*16), 128, 0, stream>>>(
      positions, senders, receivers, W0, b0, W1, b1, W2, b2,
      cheb, mY, h0, hw, gw, out);
}
